// EdgeAndNodeEmbeddingLayer_14826227105798
// MI455X (gfx1250) — matrix operand order repaired and checked by reading, not compiled, not run on hardware
//
#include <hip/hip_runtime.h>

typedef __attribute__((ext_vector_type(16))) _Float16 v16h;
typedef __attribute__((ext_vector_type(8)))  _Float16 v8h;
typedef __attribute__((ext_vector_type(8)))  float    v8f;

#define NB   8
#define NN   512
#define NOBS 7
#define NF   64

// workspace layout
static constexpr size_t ET_HALFS  = (size_t)NB * 64 * NN;      // eT[b][k][j], f16
static constexpr size_t ET_BYTES  = ET_HALFS * 2;              // 512 KB
static constexpr size_t WFT_BYTES = 64 * 64 * 2;               // 8 KB, W_feat^T f16
// then: float norm_max

// ---------------------------------------------------------------------------
// Kernel 0: global max of norm[B*N]
// ---------------------------------------------------------------------------
__global__ __launch_bounds__(256) void k_normmax(const float* __restrict__ norm,
                                                 float* __restrict__ nmax) {
    __shared__ float red[256];
    int t = threadIdx.x;
    float m = 0.0f;                        // norm >= 1, so 0 is a safe identity
    for (int i = t; i < NB * NN; i += 256) m = fmaxf(m, norm[i]);
    red[t] = m;
    __syncthreads();
    for (int s = 128; s > 0; s >>= 1) {
        if (t < s) red[t] = fmaxf(red[t], red[t + s]);
        __syncthreads();
    }
    if (t == 0) *nmax = red[0];
}

// ---------------------------------------------------------------------------
// Kernel 1: per-node edge embedding  e[b,j,k] = relu(W_edge[0,k] + nf[b,j,:]·W_edge[1:,k])
// stored transposed as f16: eT[b][k][j]  (k = 63 zero-padded)
// ---------------------------------------------------------------------------
__global__ __launch_bounds__(256) void k_embed(const float* __restrict__ nf,
                                               const float* __restrict__ W_edge,
                                               _Float16* __restrict__ eT) {
    int idx = blockIdx.x * 256 + threadIdx.x;      // B*N*64 = 2^18
    int k = idx & 63;
    int j = (idx >> 6) & (NN - 1);
    int b = idx >> 15;
    float v = 0.0f;
    if (k < 63) {
        float acc = W_edge[k];                     // adj-value coefficient (adj==1 on edges)
        const float* x = nf + ((size_t)b * NN + j) * NOBS;
#pragma unroll
        for (int f = 0; f < NOBS; ++f) acc += x[f] * W_edge[(1 + f) * 63 + k];
        v = fmaxf(acc, 0.0f);
    }
    eT[((size_t)b * 64 + k) * NN + j] = (_Float16)v;
}

// ---------------------------------------------------------------------------
// Kernel 2: W_feat^T -> f16 : WfT[n][kk] = W_feat[kk][n]
// ---------------------------------------------------------------------------
__global__ __launch_bounds__(256) void k_wft(const float* __restrict__ W_feat,
                                             _Float16* __restrict__ WfT) {
    int idx = blockIdx.x * 256 + threadIdx.x;      // 4096
    int n = idx >> 6, kk = idx & 63;
    WfT[idx] = (_Float16)W_feat[kk * 64 + n];
}

// ---------------------------------------------------------------------------
// Main kernel: per wave, one 16-row tile of one batch.
//   stage 1:  acc[b, i0..i0+15, 0..63] = adj[b, rows, :] @ e[b]      (16 WMMA K-steps x 4 col tiles)
//   stage 2:  x = acc / norm ; x[:,63] = norm/norm_max ; out = relu(x @ W_feat)  (2 K-steps x 4)
// ---------------------------------------------------------------------------
__global__ __launch_bounds__(128) void k_main(const float* __restrict__ adj,
                                              const float* __restrict__ norm,
                                              const _Float16* __restrict__ eT,
                                              const _Float16* __restrict__ WfT,
                                              const float* __restrict__ nmaxp,
                                              float* __restrict__ out) {
    __shared__ _Float16 ldsX[4][16 * 64];          // one 16x64 f16 x-tile per wave (8 KB)

    const int lane = threadIdx.x & 31;
    const int wave = threadIdx.x >> 5;
    const int tile = blockIdx.x * 4 + wave;        // 256 tiles total
    const int b    = tile >> 5;                    // 32 row-tiles per batch
    const int i0   = (tile & 31) << 4;
    const int c    = lane & 15;                    // A row / B,C column index
    const int hi   = lane >> 4;

    const float*    adjRow = adj + ((size_t)b * NN + i0 + c) * NN;
    const _Float16* eB     = eT + (size_t)b * 64 * NN;

    v8f acc[4] = {v8f{}, v8f{}, v8f{}, v8f{}};

    for (int j0 = 0; j0 < NN; j0 += 32) {
        __builtin_prefetch(adjRow + j0 + 64, 0, 0);            // stream next A data
        // A fragment: adj 16x32 tile, f32 -> f16 inline (0/1 exact)
        const v8f g0 = *(const v8f*)(adjRow + j0 + hi * 16);
        const v8f g1 = *(const v8f*)(adjRow + j0 + hi * 16 + 8);
        v16h a;
#pragma unroll
        for (int t = 0; t < 8; ++t) {
            a[t]     = (_Float16)g0[t];
            a[t + 8] = (_Float16)g1[t];
        }
        // 4 B fragments (contiguous 32B from eT), 4 WMMAs reusing A
#pragma unroll
        for (int t = 0; t < 4; ++t) {
            const v16h bf = *(const v16h*)(eB + (size_t)(t * 16 + c) * NN + j0 + hi * 16);
            acc[t] = __builtin_amdgcn_wmma_f32_16x16x32_f16(
                false, a, false, bf, (short)0, acc[t], false, false);
        }
    }

    // normalize, inject norm/norm_max into column 63, stage x-tile to LDS as f16
    const float nmax_inv = 1.0f / (*nmaxp);
    float nr[8], nrinv[8];
#pragma unroll
    for (int r = 0; r < 8; ++r) {
        nr[r]    = norm[(size_t)b * NN + i0 + r + 8 * hi];
        nrinv[r] = 1.0f / nr[r];
    }
    _Float16* X = ldsX[wave];
#pragma unroll
    for (int t = 0; t < 4; ++t) {
        const int n = t * 16 + c;
#pragma unroll
        for (int r = 0; r < 8; ++r) {
            const int   row = r + 8 * hi;
            const float xv  = (n == 63) ? (nr[r] * nmax_inv) : (acc[t][r] * nrinv[r]);
            X[row * 64 + n] = (_Float16)xv;
        }
    }
    __syncthreads();

    // second GEMM: x (16x64) @ W_feat (64x64)
    v8f acc2[4] = {v8f{}, v8f{}, v8f{}, v8f{}};
#pragma unroll
    for (int s = 0; s < 2; ++s) {
        const v8h lo = *(const v8h*)(X + c * 64 + s * 32 + hi * 16);
        const v8h hh = *(const v8h*)(X + c * 64 + s * 32 + hi * 16 + 8);
        v16h a2;
#pragma unroll
        for (int t = 0; t < 8; ++t) { a2[t] = lo[t]; a2[t + 8] = hh[t]; }
#pragma unroll
        for (int t = 0; t < 4; ++t) {
            const v16h bw = *(const v16h*)(WfT + (size_t)(t * 16 + c) * 64 + s * 32 + hi * 16);
            acc2[t] = __builtin_amdgcn_wmma_f32_16x16x32_f16(
                false, a2, false, bw, (short)0, acc2[t], false, false);
        }
    }

    // relu + store
    float* outB = out + ((size_t)b * NN + i0) * 64;
#pragma unroll
    for (int t = 0; t < 4; ++t)
#pragma unroll
        for (int r = 0; r < 8; ++r)
            outB[(r + 8 * hi) * 64 + t * 16 + c] = fmaxf(acc2[t][r], 0.0f);
}

// ---------------------------------------------------------------------------
extern "C" void kernel_launch(void* const* d_in, const int* in_sizes, int n_in,
                              void* d_out, int out_size, void* d_ws, size_t ws_size,
                              hipStream_t stream) {
    const float* node_features = (const float*)d_in[0];
    const float* adj           = (const float*)d_in[1];
    const float* norm          = (const float*)d_in[2];
    const float* W_edge        = (const float*)d_in[3];
    const float* W_feat        = (const float*)d_in[4];
    float*       out           = (float*)d_out;

    char*     ws   = (char*)d_ws;
    _Float16* eT   = (_Float16*)ws;
    _Float16* WfT  = (_Float16*)(ws + ET_BYTES);
    float*    nmax = (float*)(ws + ET_BYTES + WFT_BYTES);

    k_normmax<<<1, 256, 0, stream>>>(norm, nmax);
    k_embed<<<(NB * NN * 64) / 256, 256, 0, stream>>>(node_features, W_edge, eT);
    k_wft<<<4096 / 256, 256, 0, stream>>>(W_feat, WfT);
    k_main<<<64, 128, 0, stream>>>(adj, norm, eT, WfT, nmax, out);
}
